// EventDrivenSNN_31404800869118
// MI455X (gfx1250) — matrix operand order repaired and checked by reading, not compiled, not run on hardware
//
#include <hip/hip_runtime.h>

// ---------------- problem constants ----------------
#define B_TOT 256
#define T_TOT 4096
#define F_IN  16
#define K_IN  32      // 2F encoded features
#define H_DIM 128
#define O_DIM 2
#define MB    16      // batch tile per workgroup
#define CT    16      // timesteps staged per chunk
#define NC    (T_TOT / CT)

#define THRESH 1.0f
#define BETA   0.9f
#define ENC_TH 0.5f
#define ONE_F16 ((unsigned short)0x3C00)

typedef __attribute__((ext_vector_type(16))) _Float16 v16h;
typedef __attribute__((ext_vector_type(8)))  _Float16 v8h;
typedef __attribute__((ext_vector_type(8)))  float    v8f;
typedef int asyncv4 __attribute__((vector_size(16))); // matches builtin param

union V16 { v16h v; v8h h[2]; };

#if defined(__has_builtin)
# if __has_builtin(__builtin_amdgcn_global_load_async_to_lds_b128) && \
     __has_builtin(__builtin_amdgcn_s_wait_asynccnt)
#  define HAVE_ASYNC 1
# endif
#endif
#ifndef HAVE_ASYNC
# define HAVE_ASYNC 0
#endif

// Load a WMMA 16-bit A operand: halves 0..7 from p0, halves 8..15 from p0+16.
__device__ __forceinline__ v16h lds_a(const _Float16* p0) {
  V16 r;
  r.h[0] = *(const v8h*)p0;
  r.h[1] = *(const v8h*)(p0 + 16);
  return r.v;
}

// Load a WMMA B operand tile (halves 0..7 from p0, 8..15 from p0+16) from LDS as opaque vectors.
__device__ __forceinline__ v16h lds_b(const _Float16* p0) {
  V16 r;
  r.h[0] = *(const v8h*)p0;
  r.h[1] = *(const v8h*)(p0 + 16);
  return r.v;
}

// Stage one chunk of x (16 batch rows x CT timesteps x 16 features, fp32)
// into LDS. 1024 float4 per chunk, 4 per thread. Per batch row the chunk is
// a contiguous 1KB region of global memory; LDS layout mirrors it exactly.
__device__ __forceinline__ void stage_chunk(const float* __restrict__ x,
                                            int b0, int chunk, float* dst,
                                            int tid) {
  const int t0 = chunk * CT;
#pragma unroll
  for (int k = 0; k < 4; ++k) {
    const int i = tid + k * 256;   // float4 index within chunk
    const int b = i >> 6;          // 64 float4 per batch row
    const int w = i & 63;
    const float* gp = x + (size_t)(b0 + b) * (T_TOT * F_IN) + t0 * F_IN + w * 4;
    float* lp = dst + i * 4;
#if HAVE_ASYNC
    __builtin_amdgcn_global_load_async_to_lds_b128((asyncv4*)(void*)gp,
                                                   (asyncv4*)(void*)lp, 0, 0);
#else
    *(float4*)lp = *(const float4*)gp;
#endif
  }
}

__device__ __forceinline__ void wait_async0() {
#if HAVE_ASYNC
  __builtin_amdgcn_s_wait_asynccnt(0);
#endif
}

__global__ __launch_bounds__(256) void snn_scan_kernel(
    const float* __restrict__ x,
    const float* __restrict__ W1, const float* __restrict__ b1,
    const float* __restrict__ W2, const float* __restrict__ b2,
    const float* __restrict__ W3, const float* __restrict__ b3,
    float* __restrict__ out) {
  // Single shared block, 41KB. Persistent layout:
  //   [0,1K)   sSpkIn  [16][32] f16
  //   [1K,5K)  sSpk1   [16][128] f16
  //   [5K,9K)  sSpk2   [16][128] f16
  //   [9K,41K) sX      2 x [16][CT][16] f32 staging
  // Before the scan starts, the block doubles as an f16 weight-staging area
  // (W1:8KB + W2:32KB, then W3 padded:4KB), so per-wave WMMA B tiles can be
  // loaded as opaque ds_load_b128 vectors (genuine f16 registers).
  __shared__ __align__(16) unsigned char smem[41 * 1024];
  _Float16* sSpkIn = (_Float16*)smem;
  _Float16* sSpk1  = (_Float16*)(smem + 1024);
  _Float16* sSpk2  = (_Float16*)(smem + 5 * 1024);
  float*    sX0    = (float*)(smem + 9 * 1024);
  float*    sX1    = (float*)(smem + 25 * 1024);
  _Float16* wstage = (_Float16*)smem;   // pre-loop alias

  const int tid  = threadIdx.x;
  const int lane = tid & 31;
  const int wv   = tid >> 5;          // wave 0..7 -> hidden cols [16wv,16wv+16)
  const int b0   = blockIdx.x * MB;   // batch tile base

  // WMMA operand lane geometry (wave32, 16x16x32 f16)
  const int bn  = lane & 15;          // B operand: column N
  const int hi  = lane >> 4;          // 0/1 half-wave
  const int bk  = hi * 8;             // B operand: K sub-base (0 or 8)
  const int am  = lane & 15;          // A operand: row M
  const int akb = hi * 8;             // A operand: K sub-base (0 or 8)

  // ---- Phase A: W1,W2 -> f16 in LDS; load per-wave B tiles ----
  for (int i = tid; i < H_DIM * K_IN; i += 256)
    wstage[i] = (_Float16)W1[i];
  for (int i = tid; i < H_DIM * H_DIM; i += 256)
    wstage[H_DIM * K_IN + i] = (_Float16)W2[i];
  __syncthreads();

  V16 w1t, w2t[4], w3t[4];
  w1t.v = lds_b(wstage + (wv * 16 + bn) * K_IN + bk);
#pragma unroll
  for (int c = 0; c < 4; ++c)
    w2t[c].v = lds_b(wstage + H_DIM * K_IN + (wv * 16 + bn) * H_DIM + 32 * c + bk);
  __syncthreads();

  // ---- Phase B: W3 zero-padded to [16][128] -> f16 in LDS; load tiles ----
  for (int i = tid; i < 16 * H_DIM; i += 256) {
    const int n = i >> 7, k = i & 127;
    wstage[i] = (n < O_DIM) ? (_Float16)W3[n * H_DIM + k] : (_Float16)0.0f;
  }
  __syncthreads();
#pragma unroll
  for (int c = 0; c < 4; ++c)
    w3t[c].v = lds_b(wstage + bn * H_DIM + 32 * c + bk);
  __syncthreads();   // staging region free; x staging may overwrite

  // biases: value and value-minus-threshold (selected by reset compare)
  const float b1v = b1[wv * 16 + bn], b1r = b1v - THRESH;
  const float b2v = b2[wv * 16 + bn], b2r = b2v - THRESH;
  const float b3v = (bn < O_DIM) ? b3[bn] : 0.0f, b3r = b3v - THRESH;

  // ---- persistent state ----
  v8f m1 = {}, m2 = {}, m3 = {};     // membranes in WMMA C/D layout
  const v8f vzero = {};

  // delta-encoder: thread owns one (batch,feature) pair
  const int eb = tid >> 4;           // 0..15
  const int ef = tid & 15;           // 0..15

  // ---- stage chunk 0, prefetch chunk 1 ----
  stage_chunk(x, b0, 0, sX0, tid);
  wait_async0();
  __syncthreads();
  float ref = sX0[eb * (CT * F_IN) + ef];   // x[b,0,f]
  stage_chunk(x, b0, 1, sX1, tid);          // background

  unsigned short* usIn = (unsigned short*)sSpkIn;
  unsigned short* us1  = (unsigned short*)sSpk1;
  unsigned short* us2  = (unsigned short*)sSpk2;

  // ================== sequential scan over T ==================
  for (int t = 0; t < T_TOT; ++t) {
    const int tt    = t & (CT - 1);
    const int chunk = t >> 4;
    float* sxb      = (chunk & 1) ? sX1 : sX0;

    if (tt == 0 && t > 0) {
      wait_async0();        // this wave's loads for current buf complete
      __syncthreads();      // all waves' loads visible
      if (chunk + 1 < NC)
        stage_chunk(x, b0, chunk + 1, (chunk & 1) ? sX0 : sX1, tid);
    }

    // ---- delta encoding (t=0 row is all zeros per reference) ----
    {
      unsigned short on16 = 0, off16 = 0;
      if (t > 0) {
        const float xv = sxb[eb * (CT * F_IN) + tt * F_IN + ef];
        const float d = xv - ref;
        const bool on  = (d >=  ENC_TH);
        const bool off = (d <= -ENC_TH);
        on16  = on  ? ONE_F16 : (unsigned short)0;
        off16 = off ? ONE_F16 : (unsigned short)0;
        ref = (on | off) ? xv : ref;
      }
      usIn[eb * K_IN + ef]      = on16;
      usIn[eb * K_IN + 16 + ef] = off16;
    }
    __syncthreads();  // B1: input spikes ready

    // ---- layer 1: cur1 = spk_in @ W1^T + b1 ; LIF ----
    {
      v8f c1;
#pragma unroll
      for (int r = 0; r < 8; ++r) {
        const float m = m1[r];
        c1[r] = fmaf(BETA, m, (m > THRESH) ? b1r : b1v);
      }
      const v16h a = lds_a(&sSpkIn[am * K_IN + akb]);
      c1 = __builtin_amdgcn_wmma_f32_16x16x32_f16(false, a, false, w1t.v,
                                                  (short)0, c1, false, false);
      m1 = c1;
#pragma unroll
      for (int r = 0; r < 8; ++r)
        us1[(r + 8 * hi) * H_DIM + wv * 16 + bn] =
            (c1[r] > THRESH) ? ONE_F16 : (unsigned short)0;
    }
    __syncthreads();  // B2: spk1 ready

    // ---- layer 2: cur2 = spk1 @ W2^T + b2 ; LIF ----
    // Two independent accumulator chains (chunks 0,2 / 1,3) so the XDL pipe
    // overlaps with the ds_load/wait of the next A tiles.
    {
      v8f cA, cB;
#pragma unroll
      for (int r = 0; r < 8; ++r) {
        const float m = m2[r];
        cA[r] = fmaf(BETA, m, (m > THRESH) ? b2r : b2v);
      }
      const v16h a0 = lds_a(&sSpk1[am * H_DIM +  0 + akb]);
      const v16h a1 = lds_a(&sSpk1[am * H_DIM + 32 + akb]);
      cA = __builtin_amdgcn_wmma_f32_16x16x32_f16(false, a0, false, w2t[0].v,
                                                  (short)0, cA, false, false);
      cB = __builtin_amdgcn_wmma_f32_16x16x32_f16(false, a1, false, w2t[1].v,
                                                  (short)0, vzero, false, false);
      const v16h a2 = lds_a(&sSpk1[am * H_DIM + 64 + akb]);
      const v16h a3 = lds_a(&sSpk1[am * H_DIM + 96 + akb]);
      cA = __builtin_amdgcn_wmma_f32_16x16x32_f16(false, a2, false, w2t[2].v,
                                                  (short)0, cA, false, false);
      cB = __builtin_amdgcn_wmma_f32_16x16x32_f16(false, a3, false, w2t[3].v,
                                                  (short)0, cB, false, false);
      v8f c2;
#pragma unroll
      for (int r = 0; r < 8; ++r) c2[r] = cA[r] + cB[r];
      m2 = c2;
#pragma unroll
      for (int r = 0; r < 8; ++r)
        us2[(r + 8 * hi) * H_DIM + wv * 16 + bn] =
            (c2[r] > THRESH) ? ONE_F16 : (unsigned short)0;
    }
    __syncthreads();  // B3: spk2 ready

    // ---- layer 3 (O=2, zero-padded to N=16): wave 0 only ----
    if (wv == 0) {
      v8f cA, cB;
#pragma unroll
      for (int r = 0; r < 8; ++r) {
        const float m = m3[r];
        cA[r] = fmaf(BETA, m, (m > THRESH) ? b3r : b3v);
      }
      const v16h a0 = lds_a(&sSpk2[am * H_DIM +  0 + akb]);
      const v16h a1 = lds_a(&sSpk2[am * H_DIM + 32 + akb]);
      cA = __builtin_amdgcn_wmma_f32_16x16x32_f16(false, a0, false, w3t[0].v,
                                                  (short)0, cA, false, false);
      cB = __builtin_amdgcn_wmma_f32_16x16x32_f16(false, a1, false, w3t[1].v,
                                                  (short)0, vzero, false, false);
      const v16h a2 = lds_a(&sSpk2[am * H_DIM + 64 + akb]);
      const v16h a3 = lds_a(&sSpk2[am * H_DIM + 96 + akb]);
      cA = __builtin_amdgcn_wmma_f32_16x16x32_f16(false, a2, false, w3t[2].v,
                                                  (short)0, cA, false, false);
      cB = __builtin_amdgcn_wmma_f32_16x16x32_f16(false, a3, false, w3t[3].v,
                                                  (short)0, cB, false, false);
      v8f c3;
#pragma unroll
      for (int r = 0; r < 8; ++r) c3[r] = cA[r] + cB[r];
      m3 = c3;
      if (bn < O_DIM) {
#pragma unroll
        for (int r = 0; r < 8; ++r)
          out[((size_t)t * B_TOT + b0 + r + 8 * hi) * O_DIM + bn] = c3[r];
      }
    }
  }
}

extern "C" void kernel_launch(void* const* d_in, const int* in_sizes, int n_in,
                              void* d_out, int out_size, void* d_ws, size_t ws_size,
                              hipStream_t stream) {
  const float* x  = (const float*)d_in[0];
  const float* W1 = (const float*)d_in[1];
  const float* b1 = (const float*)d_in[2];
  const float* W2 = (const float*)d_in[3];
  const float* b2 = (const float*)d_in[4];
  const float* W3 = (const float*)d_in[5];
  const float* b3 = (const float*)d_in[6];
  float* out = (float*)d_out;

  dim3 grid(B_TOT / MB);  // 16 workgroups, one 16-row batch tile each
  dim3 block(256);        // 8 waves: one hidden-column tile per wave
  snn_scan_kernel<<<grid, block, 0, stream>>>(x, W1, b1, W2, b2, W3, b3, out);
}
